// MMDLoss_65609920414183
// MI455X (gfx1250) — matrix operand order repaired and checked by reading, not compiled, not run on hardware
//
#include <hip/hip_runtime.h>
#include <hip/hip_bf16.h>

typedef __attribute__((ext_vector_type(16))) __bf16 v16bf;
typedef __attribute__((ext_vector_type(8)))  float  v8f;

#define DDIM 256      // feature dimension (compile-time)
#define BT   128      // block tile: 128 rows of P x 128 rows of Q
#define KS   32       // K-step per WMMA (bf16 16x16x32)
#define LDA  40       // padded LDS row stride (elements) to spread banks

// Fused Gram-RBF partial-sum kernel.
// blockIdx.z: 0 -> XX, 1 -> YY, 2 -> XY
// Each 256-thread block (8 wave32) computes a 128x128 tile of one Gram
// matrix via bf16 hi/lo split WMMA (fp32-accurate), applies exp(-d2/sigma),
// and writes one deterministic partial sum.
__global__ __launch_bounds__(256)
void mmd_gram_kernel(const float* __restrict__ X, const float* __restrict__ Y,
                     const float* __restrict__ sigmaPtr,
                     float* __restrict__ partial, int N, int M)
{
    __shared__ __bf16 Ahi[BT * LDA];
    __shared__ __bf16 Alo[BT * LDA];
    __shared__ __bf16 Bhi[BT * LDA];
    __shared__ __bf16 Blo[BT * LDA];
    __shared__ float  pn[BT];
    __shared__ float  qn[BT];
    __shared__ float  red[256];

    const int z  = blockIdx.z;
    const float* P = (z == 1) ? Y : X;   // XX,XY use X rows; YY uses Y rows
    const float* Q = (z == 0) ? X : Y;   // XX uses X cols; YY,XY use Y cols
    const int ib = blockIdx.y * BT;
    const int jb = blockIdx.x * BT;
    const int t  = threadIdx.x;

    // ---- exact fp32 row norms for the 128+128 rows this block touches ----
    {
        const float* base = (t < BT) ? (P + (size_t)(ib + t) * DDIM)
                                     : (Q + (size_t)(jb + (t - BT)) * DDIM);
        const float4* b4 = (const float4*)base;
        float s = 0.0f;
        #pragma unroll 8
        for (int i = 0; i < DDIM / 4; ++i) {
            float4 v = b4[i];
            s += v.x * v.x + v.y * v.y + v.z * v.z + v.w * v.w;
        }
        if (t < BT) pn[t] = s; else qn[t - BT] = s;
    }

    const int wave = t >> 5;
    const int lane = t & 31;
    const int half = lane >> 4;   // 0: lanes 0-15, 1: lanes 16-31
    const int mn   = lane & 15;

    const v8f vzero = {0.f, 0.f, 0.f, 0.f, 0.f, 0.f, 0.f, 0.f};
    v8f c[8];
    #pragma unroll
    for (int i = 0; i < 8; ++i) c[i] = vzero;

    // ---- main K loop: stage fp32 -> bf16 hi/lo in LDS, then WMMA ----
    for (int k0 = 0; k0 < DDIM; k0 += KS) {
        __syncthreads();   // previous compute done (and norms visible)
        for (int idx = t; idx < BT * KS; idx += 256) {
            const int r = idx >> 5;
            const int k = idx & (KS - 1);
            float  v  = P[(size_t)(ib + r) * DDIM + k0 + k];
            __bf16 vh = (__bf16)v;
            Ahi[r * LDA + k] = vh;
            Alo[r * LDA + k] = (__bf16)(v - (float)vh);
            float  u  = Q[(size_t)(jb + r) * DDIM + k0 + k];
            __bf16 uh = (__bf16)u;
            Bhi[r * LDA + k] = uh;
            Blo[r * LDA + k] = (__bf16)(u - (float)uh);
        }
        __syncthreads();

        // A fragment (16x32 bf16): lane holds row m = wave*16 + (lane&15);
        // elements 0..7  = K(8*half + 0..7), elements 8..15 = K(16 + 8*half + 0..7)
        v16bf ahi, alo;
        {
            const __bf16* rh = &Ahi[(wave * 16 + mn) * LDA];
            const __bf16* rl = &Alo[(wave * 16 + mn) * LDA];
            const int kb = 8 * half;
            #pragma unroll
            for (int e = 0; e < 8; ++e) {
                ahi[e]     = rh[kb + e];
                ahi[8 + e] = rh[16 + kb + e];
                alo[e]     = rl[kb + e];
                alo[8 + e] = rl[16 + kb + e];
            }
        }

        // 8 column tiles of 16; B fragment (32x16 bf16): lane holds column
        // n = lane&15, elements 0..7 = K(8*half + 0..7), 8..15 = K(16 + 8*half + 0..7).
        #pragma unroll
        for (int tt = 0; tt < 8; ++tt) {
            v16bf bhi, blo;
            const __bf16* rh = &Bhi[(tt * 16 + mn) * LDA + 8 * half];
            const __bf16* rl = &Blo[(tt * 16 + mn) * LDA + 8 * half];
            #pragma unroll
            for (int e = 0; e < 16; ++e) { bhi[e] = rh[e < 8 ? e : 8 + e]; blo[e] = rl[e < 8 ? e : 8 + e]; }

            c[tt] = __builtin_amdgcn_wmma_f32_16x16x32_bf16(
                        false, ahi, false, bhi, (short)0, c[tt], false, false);
            c[tt] = __builtin_amdgcn_wmma_f32_16x16x32_bf16(
                        false, ahi, false, blo, (short)0, c[tt], false, false);
            c[tt] = __builtin_amdgcn_wmma_f32_16x16x32_bf16(
                        false, alo, false, bhi, (short)0, c[tt], false, false);
        }
    }

    // ---- epilogue: d2 = |x|^2 + |y|^2 - 2*dot -> exp(-d2/sigma), reduce ----
    const float ninv = -1.0f / sigmaPtr[0];
    float s = 0.0f;
    #pragma unroll
    for (int tt = 0; tt < 8; ++tt) {
        const float q2 = qn[tt * 16 + mn];
        #pragma unroll
        for (int r = 0; r < 8; ++r) {
            // C layout: VGPR r -> (m = r + 8*half, n = lane&15)
            float d2 = pn[wave * 16 + r + 8 * half] + q2 - 2.0f * c[tt][r];
            d2 = fmaxf(d2, 0.0f);
            s += __expf(d2 * ninv);
        }
    }

    red[t] = s;
    __syncthreads();
    #pragma unroll
    for (int off = 128; off > 0; off >>= 1) {
        if (t < off) red[t] += red[t + off];
        __syncthreads();
    }
    if (t == 0) {
        float denom = (z == 0) ? (float)N * (float)N
                    : (z == 1) ? (float)M * (float)M
                               : (float)N * (float)M;
        float w = (z == 2) ? -2.0f : 1.0f;
        partial[(z * gridDim.y + blockIdx.y) * gridDim.x + blockIdx.x] =
            red[0] * (w / denom);
    }
}

// Deterministic final reduction: fixed-order strided sums + fixed tree.
__global__ __launch_bounds__(256)
void mmd_reduce_kernel(const float* __restrict__ partial, int n,
                       float* __restrict__ out)
{
    __shared__ float red[256];
    const int t = threadIdx.x;
    float s = 0.0f;
    for (int i = t; i < n; i += 256) s += partial[i];
    red[t] = s;
    __syncthreads();
    #pragma unroll
    for (int off = 128; off > 0; off >>= 1) {
        if (t < off) red[t] += red[t + off];
        __syncthreads();
    }
    if (t == 0) out[0] = red[0];
}

extern "C" void kernel_launch(void* const* d_in, const int* in_sizes, int n_in,
                              void* d_out, int out_size, void* d_ws, size_t ws_size,
                              hipStream_t stream)
{
    const float* X   = (const float*)d_in[0];
    const float* Y   = (const float*)d_in[1];
    const float* sig = (const float*)d_in[2];
    float* partial   = (float*)d_ws;
    float* out       = (float*)d_out;

    const int N = in_sizes[0] / DDIM;   // 4096
    const int M = in_sizes[1] / DDIM;   // 4096

    dim3 grid(M / BT, N / BT, 3);       // 32 x 32 x 3 blocks
    mmd_gram_kernel<<<grid, 256, 0, stream>>>(X, Y, sig, partial, N, M);

    const int npart = 3 * (N / BT) * (M / BT);
    mmd_reduce_kernel<<<1, 256, 0, stream>>>(partial, npart, out);
}
